// LSTMSingle_79474074845273
// MI455X (gfx1250) — matrix operand order repaired and checked by reading, not compiled, not run on hardware
//
#include <hip/hip_runtime.h>

// ---------------------------------------------------------------------------
// LSTM over ragged sequences, fused persistent-kernel implementation (gfx1250)
//   K0: prep    - build Wcat = [W_hh | W_ih] bf16, zero h-state + barriers
//   K0b: gather - x_bf16 = emb[tokens] (f32 -> bf16), once, off critical path
//   K1: lstm    - persistent, 16 WGs, grid barrier per timestep, WMMA bf16,
//                 double-buffered async global->LDS staging (ASYNCcnt)
//   K2: fc      - hT @ W_fc^T + b_fc, WMMA bf16
// ---------------------------------------------------------------------------

typedef __bf16 bf16;
typedef __attribute__((ext_vector_type(16))) __bf16 v16bf;
typedef __attribute__((ext_vector_type(8)))  __bf16 v8bf;
typedef __attribute__((ext_vector_type(8)))  float  v8f;
typedef __attribute__((ext_vector_type(4)))  int    v4i;

#define HID    512
#define BATCH  128
#define LSEQ   512
#define FOURH  2048
#define KTOT   1024          // [h | x] concatenated K for the recurrence GEMM
#define VOCAB  50000
#define NWG    16            // persistent grid size (one WG per 32 hidden units)
#define AS_STRIDE 40         // bf16 elems per LDS row (80B, staggers banks, 16B aligned)
#define GS_STRIDE 128

// ---- CDNA5 async global->LDS path (ASYNCcnt), with safe fallback -----------
#if defined(__gfx1250__) && __has_builtin(__builtin_amdgcn_global_load_async_to_lds_b128)
#define HAVE_ASYNC 1
#else
#define HAVE_ASYNC 0
#endif

#define AS1 __attribute__((address_space(1)))
#define AS3 __attribute__((address_space(3)))

__device__ __forceinline__ void async_cp16(bf16* lds, const bf16* g) {
#if HAVE_ASYNC
    // GLOBAL_LOAD_ASYNC_TO_LDS_B128: (global int4* src, local int4* dst, ioff, cpol)
    __builtin_amdgcn_global_load_async_to_lds_b128(
        (AS1 v4i*)(g), (AS3 v4i*)(lds), /*offset=*/0, /*cpol=*/0);
#else
    *reinterpret_cast<v8bf*>(lds) = *reinterpret_cast<const v8bf*>(g);
#endif
}

#if HAVE_ASYNC
#if __has_builtin(__builtin_amdgcn_s_wait_asynccnt)
#define WAIT_ASYNCCNT(N) __builtin_amdgcn_s_wait_asynccnt(N)
#else
#define WAIT_ASYNCCNT(N) asm volatile("s_wait_asynccnt " #N ::: "memory")
#endif
#else
#define WAIT_ASYNCCNT(N) do {} while (0)
#endif

// ---------------------------------------------------------------------------

__device__ __forceinline__ v16bf ld_frag(const bf16* p0, const bf16* p1) {
    v8bf lo = *reinterpret_cast<const v8bf*>(p0);
    v8bf hi = *reinterpret_cast<const v8bf*>(p1);
    v16bf r;
#pragma unroll
    for (int i = 0; i < 8; ++i) { r[i] = lo[i]; r[i + 8] = hi[i]; }
    return r;
}

__device__ __forceinline__ v8f wmma_bf16(v16bf a, v16bf b, v8f c) {
    // D(16x16,f32) = A(16x32,bf16) * B(32x16,bf16) + C
    return __builtin_amdgcn_wmma_f32_16x16x32_bf16(
        /*neg_a=*/false, a, /*neg_b=*/false, b,
        /*c_mod=*/(short)0, c, /*reuse_a=*/false, /*reuse_b=*/false);
}

__device__ __forceinline__ float sigmoidf_fast(float x) {
    return 1.0f / (1.0f + __expf(-x));
}

__device__ __forceinline__ void grid_barrier(unsigned* cnt, unsigned* gen, unsigned nwg) {
    __syncthreads();
    if (threadIdx.x == 0) {
        __threadfence();
        unsigned g = atomicAdd(gen, 0u);            // read current generation
        unsigned arrived = atomicAdd(cnt, 1u) + 1u; // arrive
        if (arrived == nwg) {
            atomicExch(cnt, 0u);
            __threadfence();
            atomicAdd(gen, 1u);                     // release
        } else {
            while (atomicAdd(gen, 0u) == g) { __builtin_amdgcn_s_sleep(2); }
        }
        __threadfence();
    }
    __syncthreads();
}

// ---------------------------------------------------------------------------
// K0: prep — Wcat[n][k] = (k<512 ? W_hh[n][k] : W_ih[n][k-512]) in bf16,
//            zero h-state (bf16), zero barrier counters.
// ---------------------------------------------------------------------------
#define WCAT_ELEMS (FOURH * KTOT)                  // 2'097'152
#define XBF_ELEMS  ((size_t)BATCH * LSEQ * HID)    // 33'554'432
#define HST_ELEMS  (BATCH * HID)                   // 65'536
#define PREP_TOTAL (WCAT_ELEMS + HST_ELEMS + 2)

__global__ void lstm_prep_kernel(const float* __restrict__ W_ih,
                                 const float* __restrict__ W_hh,
                                 bf16* __restrict__ Wcat,
                                 bf16* __restrict__ hstate,
                                 unsigned* __restrict__ cnts) {
    int i = blockIdx.x * 256 + threadIdx.x;
    if (i < WCAT_ELEMS) {
        int n = i >> 10;          // 0..2047
        int k = i & 1023;         // 0..1023
        float v = (k < HID) ? W_hh[n * HID + k] : W_ih[n * HID + (k - HID)];
        Wcat[i] = (bf16)v;
    } else if (i < WCAT_ELEMS + HST_ELEMS) {
        hstate[i - WCAT_ELEMS] = (bf16)0.0f;
    } else if (i < PREP_TOTAL) {
        cnts[i - (WCAT_ELEMS + HST_ELEMS)] = 0u;
    }
}

// ---------------------------------------------------------------------------
// K0b: gather — x_bf16[b][l][k] = bf16(emb[tokens[b][l]][k]); 8 elems/thread.
// ---------------------------------------------------------------------------
__global__ void lstm_gather_kernel(const int* __restrict__ tokens,
                                   const float* __restrict__ emb,
                                   bf16* __restrict__ xbf) {
    size_t i = (size_t)blockIdx.x * 256 + threadIdx.x;   // over XBF_ELEMS/8
    if (i >= XBF_ELEMS / 8) return;
    size_t base = i * 8;
    int    k    = (int)(base & (HID - 1));
    size_t bl   = base / HID;                // b*L + l
    int    tok  = tokens[bl];
    const float* s = emb + (size_t)tok * HID + k;
#pragma unroll
    for (int j = 0; j < 8; ++j) xbf[base + j] = (bf16)s[j];
}

// ---------------------------------------------------------------------------
// K1: persistent LSTM recurrence.
//   WG w owns hidden units j in [32w, 32w+32); its 128 gate columns are
//   rows {g*512 + 32w + jj : g=0..3, jj=0..31} of Wcat.
//   Per step: gates = [h_bf16 | x_bf16[:,t,:]] @ Wcat_slice^T (WMMA, K=1024),
//   double-buffered async global->LDS staging, then elementwise cell update
//   (c resident in LDS), h -> global bf16, grid barrier.
// ---------------------------------------------------------------------------
__global__ __launch_bounds__(256, 1) void lstm_persistent_kernel(
    const int*   __restrict__ lengths,  // [B]
    const float* __restrict__ b_ih,     // [4H]
    const float* __restrict__ b_hh,     // [4H]
    const bf16*  __restrict__ Wcat,     // [4H, 1024] bf16
    const bf16*  __restrict__ xbf,      // [B, L, H] bf16
    bf16*        __restrict__ hstate,   // [B, H] bf16 (zeroed by prep)
    unsigned*    __restrict__ bar_cnt,
    unsigned*    __restrict__ bar_gen) {

    __shared__ __align__(16) bf16  As[2][128 * AS_STRIDE];    // A chunks [128 x 32]
    __shared__ __align__(16) bf16  Bs[2][128 * AS_STRIDE];    // B chunks [128 x 32]
    __shared__ __align__(16) float gatesS[128 * GS_STRIDE];   // pre-activations
    __shared__ __align__(16) float cS[BATCH * 32];            // resident cell state slice

    const int tid    = threadIdx.x;
    const int lane   = tid & 31;
    const int wid    = tid >> 5;
    const int wave_m = wid >> 2;          // 0..1  (rows 0-63 / 64-127)
    const int gate   = wid & 3;           // 0..3  (i, f, g, o)
    const int j0     = blockIdx.x * 32;   // hidden slice base

    // zero resident cell state
    for (int i = tid; i < BATCH * 32; i += 256) cS[i] = 0.0f;

    // staging assignment: 2 threads per row, 16 contiguous bf16 each
    const int srow = tid >> 1;
    const int skk  = (tid & 1) * 16;
    const int nglob = ((srow >> 5) * HID) + j0 + (srow & 31);  // Wcat row for Bs row
    const bf16* wrow = Wcat + (size_t)nglob * KTOT;
    const int khalf = lane >> 4;

    for (int t = 0; t < LSEQ; ++t) {
        v8f acc[4][2];
#pragma unroll
        for (int a = 0; a < 4; ++a)
#pragma unroll
            for (int b = 0; b < 2; ++b)
                acc[a][b] = (v8f){0.f, 0.f, 0.f, 0.f, 0.f, 0.f, 0.f, 0.f};

        // 4 async b128 issues per thread per chunk (uniform -> asynccnt math)
        auto stage = [&](int kc, int buf) {
            const int kg = kc * 32;
            const bf16* asrc = (kg < HID)
                ? hstate + (size_t)srow * HID + kg + skk
                : xbf + ((size_t)srow * LSEQ + t) * HID + (kg - HID) + skk;
            async_cp16(&As[buf][srow * AS_STRIDE + skk],     asrc);
            async_cp16(&As[buf][srow * AS_STRIDE + skk + 8], asrc + 8);
            const bf16* bsrc = wrow + kg + skk;
            async_cp16(&Bs[buf][srow * AS_STRIDE + skk],     bsrc);
            async_cp16(&Bs[buf][srow * AS_STRIDE + skk + 8], bsrc + 8);
        };

        stage(0, 0);                                     // pipeline prologue

        for (int kc = 0; kc < KTOT / 32; ++kc) {
            const int cur = kc & 1;
            if (kc + 1 < KTOT / 32) {
                stage(kc + 1, cur ^ 1);                  // prefetch next chunk
                WAIT_ASYNCCNT(4);                        // chunk kc landed
            } else {
                WAIT_ASYNCCNT(0);
            }
            __syncthreads();                             // cross-wave visibility

            // ---- fragments (ISA 7.12.2 layouts) + 8 WMMAs per wave ----
            v16bf afrag[4], bfrag[2];
#pragma unroll
            for (int tm = 0; tm < 4; ++tm) {
                int row = wave_m * 64 + tm * 16 + (lane & 15);
                afrag[tm] = ld_frag(&As[cur][row * AS_STRIDE + 8 * khalf],
                                    &As[cur][row * AS_STRIDE + 16 + 8 * khalf]);
            }
#pragma unroll
            for (int tn = 0; tn < 2; ++tn) {
                int col = gate * 32 + tn * 16 + (lane & 15);
                bfrag[tn] = ld_frag(&Bs[cur][col * AS_STRIDE + 8 * khalf],
                                    &Bs[cur][col * AS_STRIDE + 16 + 8 * khalf]);
            }
#pragma unroll
            for (int tm = 0; tm < 4; ++tm)
#pragma unroll
                for (int tn = 0; tn < 2; ++tn)
                    acc[tm][tn] = wmma_bf16(afrag[tm], bfrag[tn], acc[tm][tn]);
            __syncthreads();                             // buffer safe to restage
        }

        // ---- spill pre-activations to LDS (C-layout: row = r + 8*(lane>>4)) ----
#pragma unroll
        for (int tm = 0; tm < 4; ++tm)
#pragma unroll
            for (int tn = 0; tn < 2; ++tn) {
                int col = gate * 32 + tn * 16 + (lane & 15);
#pragma unroll
                for (int r = 0; r < 8; ++r) {
                    int row = wave_m * 64 + tm * 16 + r + 8 * (lane >> 4);
                    gatesS[row * GS_STRIDE + col] = acc[tm][tn][r];
                }
            }
        __syncthreads();

        // ---- elementwise LSTM cell update (validity-masked freeze) ----
        for (int p = tid; p < BATCH * 32; p += 256) {
            int row = p >> 5;
            int jj  = p & 31;
            int j   = j0 + jj;
            float gi = gatesS[row * GS_STRIDE + jj]       + b_ih[j]          + b_hh[j];
            float gf = gatesS[row * GS_STRIDE + 32 + jj]  + b_ih[HID + j]    + b_hh[HID + j];
            float gg = gatesS[row * GS_STRIDE + 64 + jj]  + b_ih[2*HID + j]  + b_hh[2*HID + j];
            float go = gatesS[row * GS_STRIDE + 96 + jj]  + b_ih[3*HID + j]  + b_hh[3*HID + j];
            gi = sigmoidf_fast(gi);
            gf = sigmoidf_fast(gf);
            gg = tanhf(gg);
            go = sigmoidf_fast(go);
            float c_old = cS[p];
            float c_new = gf * c_old + gi * gg;
            float h_new = go * tanhf(c_new);
            if (t < lengths[row]) {          // freeze state past sequence length
                cS[p] = c_new;
                hstate[(size_t)row * HID + j] = (bf16)h_new;
            }
        }

        grid_barrier(bar_cnt, bar_gen, NWG); // h fully published before next step
    }
}

// ---------------------------------------------------------------------------
// K2: out[B,V] = hT @ W_fc^T + b_fc   (M=128, N=50000, K=512; WMMA bf16)
// ---------------------------------------------------------------------------
__global__ __launch_bounds__(256) void lstm_fc_kernel(
    const bf16*  __restrict__ hstate,  // [B, H] bf16
    const float* __restrict__ W_fc,    // [V, H] f32
    const float* __restrict__ b_fc,    // [V]
    float*       __restrict__ out) {   // [B, V]

    __shared__ __align__(16) bf16 As[128 * AS_STRIDE];
    __shared__ __align__(16) bf16 Bs[128 * AS_STRIDE];

    const int tid    = threadIdx.x;
    const int lane   = tid & 31;
    const int wid    = tid >> 5;
    const int wave_m = wid >> 2;
    const int wave_n = wid & 3;
    const int n0     = blockIdx.x * 128;
    const int khalf  = lane >> 4;

    const int srow = tid >> 1;
    const int skk  = (tid & 1) * 16;
    const int nglob = n0 + srow;

    v8f acc[4][2];
#pragma unroll
    for (int a = 0; a < 4; ++a)
#pragma unroll
        for (int b = 0; b < 2; ++b)
            acc[a][b] = (v8f){0.f, 0.f, 0.f, 0.f, 0.f, 0.f, 0.f, 0.f};

    for (int kc = 0; kc < HID / 32; ++kc) {
        const int kg = kc * 32;
        // A: final hidden state, already bf16
        {
            const bf16* src = hstate + (size_t)srow * HID + kg + skk;
            *reinterpret_cast<v8bf*>(&As[srow * AS_STRIDE + skk])     =
                *reinterpret_cast<const v8bf*>(src);
            *reinterpret_cast<v8bf*>(&As[srow * AS_STRIDE + skk + 8]) =
                *reinterpret_cast<const v8bf*>(src + 8);
        }
        // B: W_fc rows (f32 -> bf16), zero-pad past V
        if (nglob < VOCAB) {
            const float* src = W_fc + (size_t)nglob * HID + kg + skk;
#pragma unroll
            for (int j = 0; j < 16; ++j)
                Bs[srow * AS_STRIDE + skk + j] = (bf16)src[j];
        } else {
#pragma unroll
            for (int j = 0; j < 16; ++j)
                Bs[srow * AS_STRIDE + skk + j] = (bf16)0.0f;
        }
        __syncthreads();

        v16bf afrag[4], bfrag[2];
#pragma unroll
        for (int tm = 0; tm < 4; ++tm) {
            int row = wave_m * 64 + tm * 16 + (lane & 15);
            afrag[tm] = ld_frag(&As[row * AS_STRIDE + 8 * khalf],
                                &As[row * AS_STRIDE + 16 + 8 * khalf]);
        }
#pragma unroll
        for (int tn = 0; tn < 2; ++tn) {
            int col = wave_n * 32 + tn * 16 + (lane & 15);
            bfrag[tn] = ld_frag(&Bs[col * AS_STRIDE + 8 * khalf],
                                &Bs[col * AS_STRIDE + 16 + 8 * khalf]);
        }
#pragma unroll
        for (int tm = 0; tm < 4; ++tm)
#pragma unroll
            for (int tn = 0; tn < 2; ++tn)
                acc[tm][tn] = wmma_bf16(afrag[tm], bfrag[tn], acc[tm][tn]);
        __syncthreads();
    }

    // store with bias, bounds-checked on N (V = 50000 is not a tile multiple)
#pragma unroll
    for (int tm = 0; tm < 4; ++tm)
#pragma unroll
        for (int tn = 0; tn < 2; ++tn) {
            int col = n0 + wave_n * 32 + tn * 16 + (lane & 15);
            if (col < VOCAB) {
                float bias = b_fc[col];
#pragma unroll
                for (int r = 0; r < 8; ++r) {
                    int row = wave_m * 64 + tm * 16 + r + 8 * (lane >> 4);
                    out[(size_t)row * VOCAB + col] = acc[tm][tn][r] + bias;
                }
            }
        }
}

// ---------------------------------------------------------------------------

extern "C" void kernel_launch(void* const* d_in, const int* in_sizes, int n_in,
                              void* d_out, int out_size, void* d_ws, size_t ws_size,
                              hipStream_t stream) {
    const int*   tokens  = (const int*)d_in[0];
    const int*   lengths = (const int*)d_in[1];
    const float* emb     = (const float*)d_in[2];
    const float* W_ih    = (const float*)d_in[3];
    const float* W_hh    = (const float*)d_in[4];
    const float* b_ih    = (const float*)d_in[5];
    const float* b_hh    = (const float*)d_in[6];
    const float* W_fc    = (const float*)d_in[7];
    const float* b_fc    = (const float*)d_in[8];
    float* out = (float*)d_out;

    char* ws = (char*)d_ws;
    bf16*     Wcat   = (bf16*)ws;                                        // 4 MiB
    bf16*     xbf    = (bf16*)(ws + (size_t)WCAT_ELEMS * 2);             // 64 MiB
    bf16*     hstate = (bf16*)(ws + (size_t)WCAT_ELEMS * 2 + XBF_ELEMS * 2);
    unsigned* cnts   = (unsigned*)(ws + (size_t)WCAT_ELEMS * 2 + XBF_ELEMS * 2
                                      + (size_t)HST_ELEMS * 2);

    // K0: build bf16 weight block, zero state + barrier counters
    lstm_prep_kernel<<<(PREP_TOTAL + 255) / 256, 256, 0, stream>>>(
        W_ih, W_hh, Wcat, hstate, cnts);

    // K0b: embedding gather + bf16 convert, once
    lstm_gather_kernel<<<(unsigned)((XBF_ELEMS / 8 + 255) / 256), 256, 0, stream>>>(
        tokens, emb, xbf);

    // K1: persistent recurrence (16 WGs, grid barrier per timestep)
    lstm_persistent_kernel<<<NWG, 256, 0, stream>>>(
        lengths, b_ih, b_hh, Wcat, xbf, hstate, cnts, cnts + 1);

    // K2: final projection to vocabulary
    lstm_fc_kernel<<<(VOCAB + 127) / 128, 256, 0, stream>>>(
        hstate, W_fc, b_fc, out);
}
